// XnorActivationConv2d_5428838662863
// MI455X (gfx1250) — 8-bit matrix operand order repaired and checked by reading, compiled, not run on hardware
//
#include <hip/hip_runtime.h>
#include <stdint.h>


#define N_IMG 32
#define C_IN  256
#define H_    56
#define W_    56
#define K_OUT 256
#define PIX   (H_ * W_)
#define WCNT  (K_OUT * C_IN * 9)
#define NPIX  (N_IMG * PIX)

#define ALPHA_OFF 0
#define ZERO_OFF  64
#define PART_OFF  1024
#define A_OFF     (PART_OFF + 2304 * 4)
#define KA_OFF    (A_OFF + NPIX * 4)
#define WBT_OFF   (KA_OFF + NPIX * 4)
#define XS_OFF    (WBT_OFF + WCNT)

typedef __attribute__((ext_vector_type(8))) int v8i;
typedef int v4i_t __attribute__((vector_size(16)));
typedef __attribute__((address_space(1))) v4i_t gv4i;
typedef __attribute__((address_space(3))) v4i_t lv4i;
typedef unsigned int u32x4 __attribute__((vector_size(16)));
typedef int i32x8 __attribute__((vector_size(32)));
typedef int i32x4 __attribute__((vector_size(16)));

#if defined(__has_builtin)
#if __has_builtin(__builtin_amdgcn_global_load_async_to_lds_b128)
#define XNOR_ASYNC_LDS 1
#endif
#if __has_builtin(__builtin_amdgcn_s_wait_asynccnt)
#define XNOR_WAIT_ASYNC_BUILTIN 1
#endif
#if __has_builtin(__builtin_amdgcn_tensor_load_to_lds)
#define XNOR_TDM 1
#endif
#if __has_builtin(__builtin_amdgcn_s_wait_tensorcnt)
#define XNOR_WAIT_TENSOR_BUILTIN 1
#endif
#endif

#ifdef XNOR_TDM
#define B_ASYNC_OPS 0
#else
#define B_ASYNC_OPS 4
#endif
#define STEP_ASYNC_OPS (1 + B_ASYNC_OPS)

__device__ __forceinline__ void async_copy16(const void* g, void* l) {
#ifdef XNOR_ASYNC_LDS
  __builtin_amdgcn_global_load_async_to_lds_b128((gv4i*)(g), (lv4i*)(l), 0, 0);
#else
  *(int4*)l = *(const int4*)g;
#endif
}

__device__ __forceinline__ void wait_async_pipe(bool next_inflight) {
#ifdef XNOR_ASYNC_LDS
#ifdef XNOR_WAIT_ASYNC_BUILTIN
  if (next_inflight) __builtin_amdgcn_s_wait_asynccnt(STEP_ASYNC_OPS);
  else               __builtin_amdgcn_s_wait_asynccnt(0);
#else
  if (next_inflight) asm volatile("s_wait_asynccnt %0" :: "i"(STEP_ASYNC_OPS) : "memory");
  else               asm volatile("s_wait_asynccnt 0" ::: "memory");
#endif
#endif
}

__device__ __forceinline__ void wait_tensor_pipe(bool next_inflight) {
#ifdef XNOR_TDM
#ifdef XNOR_WAIT_TENSOR_BUILTIN
  if (next_inflight) __builtin_amdgcn_s_wait_tensorcnt(1);
  else               __builtin_amdgcn_s_wait_tensorcnt(0);
#else
  if (next_inflight) asm volatile("s_wait_tensorcnt 0x1" ::: "memory");
  else               asm volatile("s_wait_tensorcnt 0x0" ::: "memory");
#endif
#endif
}

__device__ __forceinline__ void tdm_load_b16k(const void* gsrc, void* lds_dst) {
#ifdef XNOR_TDM
  unsigned lds_off =
      (unsigned)(size_t)(__attribute__((address_space(3))) char*)lds_dst;
  unsigned long long ga = (unsigned long long)(size_t)gsrc;
  u32x4 g0 = {1u,
              lds_off,
              (unsigned)(ga & 0xffffffffu),
              (unsigned)((ga >> 32) & 0x01ffffffu) | (2u << 30)};
  i32x8 g1 = {(int)(2u << 16),
              (int)(4096u << 16),
              (int)(1u << 16),
              (int)(4096u << 16),
              1,
              4096,
              0, 0};
  i32x4 z4 = {0, 0, 0, 0};
#if __clang_major__ >= 23
  i32x8 z8 = {0, 0, 0, 0, 0, 0, 0, 0};
  __builtin_amdgcn_tensor_load_to_lds(g0, g1, z4, z4, z8, 0);
#else
  __builtin_amdgcn_tensor_load_to_lds(g0, g1, z4, z4, 0);
#endif
#else
  (void)gsrc; (void)lds_dst;
#endif
}

__global__ __launch_bounds__(256) void pack_w_kernel(
    const float* __restrict__ w, int8_t* __restrict__ wBt,
    float* __restrict__ part) {
  int idx = blockIdx.x * 256 + threadIdx.x;
  float av = 0.f;
  if (idx < WCNT) {
    float v = w[idx];
    av = fabsf(v);
    int o  = idx / (C_IN * 9);
    int c  = (idx / 9) % C_IN;
    int kk = idx % 9;
    int cc = c >> 6, cr = c & 63;
    int8_t s = (int8_t)((v > 0.f) - (v < 0.f));
    wBt[((size_t)((kk * 4 + cc) * K_OUT + o)) * 64 + cr] = s;
  }
  __shared__ float red[256];
  red[threadIdx.x] = av;
  __syncthreads();
  for (int s = 128; s > 0; s >>= 1) {
    if (threadIdx.x < s) red[threadIdx.x] += red[threadIdx.x + s];
    __syncthreads();
  }
  if (threadIdx.x == 0) part[blockIdx.x] = red[0];
}

__global__ __launch_bounds__(256) void reduce_alpha_kernel(
    const float* __restrict__ part, float* __restrict__ alpha) {
  __shared__ float red[256];
  float s = 0.f;
  for (int i = threadIdx.x; i < 2304; i += 256) s += part[i];
  red[threadIdx.x] = s;
  __syncthreads();
  for (int k = 128; k > 0; k >>= 1) {
    if (threadIdx.x < k) red[threadIdx.x] += red[threadIdx.x + k];
    __syncthreads();
  }
  if (threadIdx.x == 0) alpha[0] = red[0] * (1.f / (float)WCNT);
}

__global__ __launch_bounds__(256) void pack_x_kernel(
    const float* __restrict__ x, int8_t* __restrict__ xs,
    float* __restrict__ A) {
  int n = blockIdx.x / H_, h = blockIdx.x % H_;
  int t = threadIdx.x;
  int w = t & 63, cq = t >> 6;
  float asum = 0.f;
  if (w < W_) {
    const float* xp = x + (size_t)n * C_IN * PIX + h * W_ + w;
    int8_t* xsp = xs + ((size_t)((n * H_ + h) * W_ + w)) * C_IN;
#pragma unroll 4
    for (int c0 = 0; c0 < 64; ++c0) {
      int c = c0 * 4 + cq;
      float v = xp[(size_t)c * PIX];
      asum += fabsf(v);
      xsp[c] = (int8_t)((v > 0.f) - (v < 0.f));
    }
  }
  __shared__ float red[256];
  red[t] = asum;
  __syncthreads();
  if (cq == 0 && w < W_)
    A[(n * H_ + h) * W_ + w] =
        (red[w] + red[64 + w] + red[128 + w] + red[192 + w]) * (1.f / 256.f);
}

__global__ __launch_bounds__(64) void boxfilter_kernel(
    const float* __restrict__ A, const float* __restrict__ alpha,
    float* __restrict__ KA) {
  int n = blockIdx.x / H_, h = blockIdx.x % H_;
  int w = threadIdx.x;
  if (w >= W_) return;
  float s = 0.f;
  for (int dh = -1; dh <= 1; ++dh) {
    int hh = h + dh;
    if (hh < 0 || hh >= H_) continue;
    for (int dw = -1; dw <= 1; ++dw) {
      int ww = w + dw;
      if (ww < 0 || ww >= W_) continue;
      s += A[(n * H_ + hh) * W_ + ww];
    }
  }
  KA[(n * H_ + h) * W_ + w] = s * (1.f / 9.f) * alpha[0];
}

__global__ __launch_bounds__(256) void xnor_conv_kernel(
    const int8_t* __restrict__ xs, const int8_t* __restrict__ wBt,
    const float* __restrict__ KA, float* __restrict__ out,
    const int8_t* __restrict__ zero64) {
  __shared__ __align__(16) int8_t smem[2 * 64 * 64 + 2 * 256 * 64];

  int n = blockIdx.x / H_, h = blockIdx.x % H_;
  int t = threadIdx.x;
  int wave = t >> 5, lane = t & 31;
  int col = lane & 15, g = lane >> 4;
  int m_ld = t >> 2, seg = t & 3;

  v8i acc[4][2];
  v8i vzero = {0, 0, 0, 0, 0, 0, 0, 0};
#pragma unroll
  for (int mt = 0; mt < 4; ++mt)
#pragma unroll
    for (int nt = 0; nt < 2; ++nt) acc[mt][nt] = vzero;


  auto issueA = [&](int kk, int8_t* As) {
    int kpos = kk >> 2, cc = kk & 3;
    int dh = kpos / 3 - 1, dw = kpos % 3 - 1;
    int hh = h + dh, ww = m_ld + dw;
    bool valid = (m_ld < W_) && (hh >= 0) && (hh < H_) && (ww >= 0) && (ww < W_);
    const int8_t* gsrc =
        valid ? (xs + ((size_t)((n * H_ + hh) * W_ + ww)) * C_IN + cc * 64 + seg * 16)
              : (zero64 + seg * 16);
    async_copy16(gsrc, As + m_ld * 64 + seg * 16);
  };
  auto issueB = [&](int kk, int8_t* Bs) {
    const int8_t* gb = wBt + (size_t)kk * (K_OUT * 64);
#ifdef XNOR_TDM
    if (wave == 0) tdm_load_b16k(gb, Bs);
#else
#pragma unroll
    for (int q = 0; q < 4; ++q)
      async_copy16(gb + t * 64 + q * 16, Bs + t * 64 + q * 16);
#endif
  };
  auto compute = [&](const int8_t* As, const int8_t* Bs) {
    v8i bfrag[2];
#pragma unroll
    for (int nt = 0; nt < 2; ++nt) {
      const int8_t* bb = Bs + (wave * 32 + nt * 16 + col) * 64 + g * 16;
      int4 lo = *(const int4*)(bb);
      int4 hi = *(const int4*)(bb + 32);
      v8i b = {lo.x, lo.y, lo.z, lo.w, hi.x, hi.y, hi.z, hi.w};
      bfrag[nt] = b;
    }
#pragma unroll
    for (int mt = 0; mt < 4; ++mt) {
      const int8_t* ab = As + (mt * 16 + col) * 64 + g * 16;
      int4 p0 = *(const int4*)(ab);
      int4 p1 = *(const int4*)(ab + 32);
      v8i a = {p0.x, p0.y, p0.z, p0.w, p1.x, p1.y, p1.z, p1.w};
#pragma unroll
      for (int nt = 0; nt < 2; ++nt)
        acc[mt][nt] = __builtin_amdgcn_wmma_i32_16x16x64_iu8(
             true, a,  true, bfrag[nt],
            acc[mt][nt],  false,  false);
    }
  };

  issueB(0, smem + 8192);
  issueA(0, smem);
  for (int kk = 0; kk < 36; ++kk) {
    int cur = kk & 1, nxt = cur ^ 1;
    bool more = (kk + 1) < 36;
    if (more) {
      issueB(kk + 1, smem + 8192 + nxt * 16384);
      issueA(kk + 1, smem + nxt * 4096);
    }
    wait_async_pipe(more);
    wait_tensor_pipe(more);
    __syncthreads();
    compute(smem + cur * 4096, smem + 8192 + cur * 16384);
    __syncthreads();
  }

  float* ep = (float*)(void*)smem + wave * (32 * 20);
  const float* KArow = KA + (n * H_ + h) * W_;
#pragma unroll
  for (int mt = 0; mt < 4; ++mt) {
#pragma unroll
    for (int nt = 0; nt < 2; ++nt) {
      int o_l = nt * 16 + col;
#pragma unroll
      for (int r = 0; r < 8; ++r) {
        int m = mt * 16 + r + g * 8;
        float scale = (m < W_) ? KArow[m] : 0.f;
        ep[o_l * 20 + r + g * 8] = (float)acc[mt][nt][r] * scale;
      }
    }
#pragma unroll
    for (int it = 0; it < 4; ++it) {
      int o_l = it * 8 + (lane >> 2);
      int sg = lane & 3;
      int m0 = mt * 16 + sg * 4;
      if (m0 < W_) {
        float4 vv = *(const float4*)&ep[o_l * 20 + sg * 4];
        int o = wave * 32 + o_l;
        *(float4*)&out[(((size_t)n * K_OUT + o) * H_ + h) * W_ + m0] = vv;
      }
    }
  }
}

extern "C" void kernel_launch(void* const* d_in, const int* in_sizes, int n_in,
                              void* d_out, int out_size, void* d_ws, size_t ws_size,
                              hipStream_t stream) {
  const float* x = (const float*)d_in[0];
  const float* w = (const float*)d_in[1];
  float* out = (float*)d_out;
  char* ws = (char*)d_ws;

  (void)hipMemsetAsync(ws, 0, 256, stream);

  pack_w_kernel<<<WCNT / 256, 256, 0, stream>>>(
      w, (int8_t*)(ws + WBT_OFF), (float*)(ws + PART_OFF));
  reduce_alpha_kernel<<<1, 256, 0, stream>>>(
      (const float*)(ws + PART_OFF), (float*)(ws + ALPHA_OFF));
  pack_x_kernel<<<N_IMG * H_, 256, 0, stream>>>(
      x, (int8_t*)(ws + XS_OFF), (float*)(ws + A_OFF));
  boxfilter_kernel<<<N_IMG * H_, 64, 0, stream>>>(
      (const float*)(ws + A_OFF), (const float*)(ws + ALPHA_OFF),
      (float*)(ws + KA_OFF));
  xnor_conv_kernel<<<N_IMG * H_, 256, 0, stream>>>(
      (const int8_t*)(ws + XS_OFF), (const int8_t*)(ws + WBT_OFF),
      (const float*)(ws + KA_OFF), out, (const int8_t*)(ws + ZERO_OFF));

  (void)in_sizes; (void)n_in; (void)out_size; (void)ws_size;
}
